// TEMLite_40467181863203
// MI455X (gfx1250) — compile-verified, matrix operand order checked by reading, not run on hardware
//
#include <hip/hip_runtime.h>
#include <cmath>

// ---------------- model constants ----------------
#define Bv      256
#define Tv      256
#define N_OBS   512
#define N_ACT   8
#define N_ST    1024
#define Gd      64
#define Xd      32
#define Pd      128
#define Hd      128
#define ETA     0.4f
#define LAM     0.995f
#define KAPPA   0.8f
#define OBS_W   1.0f
#define MEM_W   0.1f
#define ST_W    1.0f

#define BT      (Bv * Tv)          // 65536 rows
#define N_TOT   (N_OBS + N_ST)     // 1536 decoder outputs
#define MSTR    132                // M row stride (floats): 528B, 16B aligned, banks offset by 4/row

// d_out layout (f32, concatenated in return order)
#define OFF_OBSL  ((size_t)0)
#define OFF_STL   ((size_t)BT * N_OBS)                   // 33,554,432
#define OFF_P     (OFF_STL + (size_t)BT * N_ST)          // 100,663,296
#define OFF_PM    (OFF_P + (size_t)BT * Pd)              // 109,051,904
#define OFF_LOSS  (OFF_PM + (size_t)BT * Pd)             // 117,440,512

typedef __attribute__((ext_vector_type(16))) __bf16        v16bf;
typedef __attribute__((ext_vector_type(8)))  float         v8f;
typedef __attribute__((ext_vector_type(4)))  unsigned int  uv4;
typedef unsigned short ushort_t;

// float -> bf16 (round to nearest even)
__device__ __forceinline__ ushort_t f2bf(float f) {
  unsigned int x = __float_as_uint(f);
  unsigned int r = x + 0x7FFFu + ((x >> 16) & 1u);
  return (ushort_t)(r >> 16);
}

__device__ __forceinline__ float sigm(float x) { return 1.0f / (1.0f + __expf(-x)); }

__device__ __forceinline__ float dot4(float4 a, float4 b) {
  return a.x * b.x + a.y * b.y + a.z * b.z + a.w * b.w;
}

// fixed-order block reductions (deterministic); blockDim must be power of 2
__device__ __forceinline__ float redSum(float v, float* red, int tid, int n) {
  red[tid] = v; __syncthreads();
  for (int s = n >> 1; s > 0; s >>= 1) {
    if (tid < s) red[tid] += red[tid + s];
    __syncthreads();
  }
  float r = red[0]; __syncthreads();
  return r;
}
__device__ __forceinline__ float redMax(float v, float* red, int tid, int n) {
  red[tid] = v; __syncthreads();
  for (int s = n >> 1; s > 0; s >>= 1) {
    if (tid < s) red[tid] = fmaxf(red[tid], red[tid + s]);
    __syncthreads();
  }
  float r = red[0]; __syncthreads();
  return r;
}

// =====================================================================
// Kernel 0: pack decoder weights [dec_obs_w ; dec_st_w] -> bf16 (1536x128)
// =====================================================================
__global__ void teml_prep_w(const float* __restrict__ ow, const float* __restrict__ sw,
                            ushort_t* __restrict__ W) {
  int i = blockIdx.x * blockDim.x + threadIdx.x;   // 0 .. 1536*128-1
  int n = i >> 7, k = i & 127;
  float v = (n < N_OBS) ? ow[n * Pd + k] : sw[(n - N_OBS) * Pd + k];
  W[i] = f2bf(v);
}

// =====================================================================
// Kernel 1: persistent recurrent kernel. One WG per batch element,
// M (128x128, row stride 132 floats -> all row ops are ds_*_b128)
// resident in LDS for the whole T loop. 128 threads/WG.
// Emits p_norm, p_mem (f32 outputs) and p_comb (bf16 scratch).
// =====================================================================
__global__ void __launch_bounds__(128)
teml_recurrent(const int* __restrict__ obs, const int* __restrict__ actions,
               const float* __restrict__ obs_emb, const float* __restrict__ act_emb,
               const float* __restrict__ g_init,
               const float* __restrict__ w_ih, const float* __restrict__ w_hh,
               const float* __restrict__ b_ih, const float* __restrict__ b_hh,
               const float* __restrict__ enc_w1, const float* __restrict__ enc_b1,
               const float* __restrict__ ln_g, const float* __restrict__ ln_b,
               const float* __restrict__ enc_w2, const float* __restrict__ enc_b2,
               float* __restrict__ p_out, float* __restrict__ pm_out,
               ushort_t* __restrict__ pcomb_out) {
  extern __shared__ __align__(16) float sm[];
  float* M   = sm;                 // 128*132 = 16896 floats (row stride 132)
  float* gb  = M  + 128 * MSTR;    // 64
  float* ab  = gb + 64;            // 64
  float* xb  = ab + 64;            // 32
  float* gi  = xb + 32;            // 192
  float* gh  = gi + 192;           // 192
  float* hb  = gh + 192;           // 128
  float* pb  = hb + 128;           // 128  (p_norm)
  float* qb  = pb + 128;           // 128
  float* tb  = qb + 128;           // 128
  float* red = tb + 128;           // 128

  const int tid = threadIdx.x;
  const int b   = blockIdx.x;

  // init M = 0 (vectorized: 128*33 float4s exactly), g = g_init
  {
    float4 z4 = make_float4(0.f, 0.f, 0.f, 0.f);
    float4* M4 = (float4*)M;
    for (int i = tid; i < 128 * 33; i += 128) M4[i] = z4;
  }
  if (tid < Gd) gb[tid] = g_init[tid];
  __syncthreads();

  for (int t = 0; t < Tv; ++t) {
    const size_t row = (size_t)b * Tv + t;

    // gather obs embedding x (32) and previous-action embedding a (64)
    int o = obs[row];
    if (tid < Xd) xb[tid] = obs_emb[(size_t)o * Xd + tid];
    if (t > 0) {
      int ac = actions[row - 1];
      if (tid < Gd) ab[tid] = act_emb[(size_t)ac * Gd + tid];
    }
    __syncthreads();

    // ---- GRU (skipped at t==0) ----
    if (t > 0) {
      const float4* ab4 = (const float4*)ab;
      const float4* gb4 = (const float4*)gb;
      for (int r = tid; r < 3 * Gd; r += 128) {
        float si = b_ih[r], sh = b_hh[r];
        const float4* wi4 = (const float4*)(w_ih + (size_t)r * Gd);  // 256B rows, aligned
        const float4* wh4 = (const float4*)(w_hh + (size_t)r * Gd);
        #pragma unroll
        for (int k = 0; k < Gd / 4; ++k) {
          si += dot4(wi4[k], ab4[k]);
          sh += dot4(wh4[k], gb4[k]);
        }
        gi[r] = si; gh[r] = sh;
      }
      __syncthreads();
      if (tid < Gd) {
        float r = sigm(gi[tid] + gh[tid]);
        float z = sigm(gi[Gd + tid] + gh[Gd + tid]);
        float n = tanhf(gi[2 * Gd + tid] + r * gh[2 * Gd + tid]);
        tb[tid] = (1.0f - z) * n + z * gb[tid];
      }
      __syncthreads();
      if (tid < Gd) gb[tid] = tb[tid];
      __syncthreads();
    }

    // ---- encoder layer 1 + LayerNorm + relu ----
    float hv = enc_b1[tid];
    {
      const float4* w14 = (const float4*)(enc_w1 + (size_t)tid * (Gd + Xd)); // 384B rows, aligned
      const float4* gb4 = (const float4*)gb;
      const float4* xb4 = (const float4*)xb;
      #pragma unroll
      for (int k = 0; k < Gd / 4; ++k) hv += dot4(w14[k], gb4[k]);
      #pragma unroll
      for (int k = 0; k < Xd / 4; ++k) hv += dot4(w14[Gd / 4 + k], xb4[k]);
    }
    float mu  = redSum(hv, red, tid, 128) * (1.0f / 128.0f);
    float d   = hv - mu;
    float var = redSum(d * d, red, tid, 128) * (1.0f / 128.0f);
    float hn  = fmaxf(d * rsqrtf(var + 1e-5f) * ln_g[tid] + ln_b[tid], 0.0f);
    hb[tid] = hn;
    __syncthreads();

    // ---- encoder layer 2 + relu + l2norm ----
    float pv = enc_b2[tid];
    {
      const float4* w24 = (const float4*)(enc_w2 + (size_t)tid * Hd);        // 512B rows, aligned
      const float4* hb4 = (const float4*)hb;
      #pragma unroll
      for (int k = 0; k < Hd / 4; ++k) pv += dot4(w24[k], hb4[k]);
    }
    pv = fmaxf(pv, 0.0f) + 1e-6f;
    float n2  = redSum(pv * pv, red, tid, 128);
    float rin = 1.0f / fmaxf(sqrtf(n2), 1e-12f);
    float pn  = pv * rin;
    pb[tid] = pn;
    qb[tid] = pn;
    p_out[row * Pd + tid] = pn;
    __syncthreads();

    // ---- Hebbian attractor retrieval: q = l2norm(KAPPA*q + M q), x3 ----
    for (int it = 0; it < 3; ++it) {
      float s = KAPPA * qb[tid];
      const float4* Mr4 = (const float4*)(M + tid * MSTR);   // b128 LDS reads
      const float4* qb4 = (const float4*)qb;
      #pragma unroll 8
      for (int k = 0; k < Pd / 4; ++k) s += dot4(Mr4[k], qb4[k]);
      tb[tid] = s;
      float qn2 = redSum(s * s, red, tid, 128);   // internal sync fences qb reads
      float qri = 1.0f / fmaxf(sqrtf(qn2), 1e-12f);
      qb[tid] = tb[tid] * qri;
      __syncthreads();
    }
    pm_out[row * Pd + tid] = qb[tid];

    // ---- p_comb = l2norm(p_norm + p_mem) -> bf16 scratch for the GEMM ----
    float c   = pb[tid] + qb[tid];
    float cn2 = redSum(c * c, red, tid, 128);
    float cri = 1.0f / fmaxf(sqrtf(cn2), 1e-12f);
    pcomb_out[row * Pd + tid] = f2bf(c * cri);

    // ---- M = LAM*M + ETA * pn pn^T (vectorized b128 RMW in LDS) ----
    {
      const float4* pb4 = (const float4*)pb;
      for (int idx = tid; idx < 128 * 32; idx += 128) {
        int i = idx >> 5, j = idx & 31;           // row i, float4-column j
        float4* mp = (float4*)(M + i * MSTR) + j;
        float4 mv = *mp;
        float4 pv4 = pb4[j];
        float  pi = pb[i];
        mv.x = LAM * mv.x + ETA * pi * pv4.x;
        mv.y = LAM * mv.y + ETA * pi * pv4.y;
        mv.z = LAM * mv.z + ETA * pi * pv4.z;
        mv.w = LAM * mv.w + ETA * pi * pv4.w;
        *mp = mv;
      }
    }
    __syncthreads();
  }
}

// =====================================================================
// Kernel 2: decoder GEMM  logits(65536 x 1536) = Pcomb(65536 x 128) * W^T
// bf16 A/B, f32 accumulate via v_wmma_f32_16x16x32_bf16.
// Block tile 128(M) x 128(N); 8 waves, each 32x64 (2x4 tiles of 16x16).
// =====================================================================
__global__ void __launch_bounds__(256)
teml_decoder_gemm(const ushort_t* __restrict__ A, const ushort_t* __restrict__ W,
                  const float* __restrict__ obs_b, const float* __restrict__ st_b,
                  float* __restrict__ out_obs, float* __restrict__ out_st) {
  extern __shared__ __align__(16) ushort_t ds[];
  ushort_t* sA = ds;                 // 128 rows x 136 halves (pad keeps 16B align, spreads banks)
  ushort_t* sB = ds + 128 * 136;

  const int tid = threadIdx.x;
  const int m0  = blockIdx.x * 128;
  const int n0  = blockIdx.y * 128;

  // stage A block (rows of p_comb) and B block (rows of W = output features)
  for (int c = tid; c < 2048; c += 256) {              // 2048 x 16B = 32KB
    int r = c >> 4, col = c & 15;
    *(uv4*)(sA + r * 136 + col * 8) = *(const uv4*)(A + (size_t)(m0 + r) * Pd + col * 8);
  }
  for (int c = tid; c < 2048; c += 256) {
    int r = c >> 4, col = c & 15;
    *(uv4*)(sB + r * 136 + col * 8) = *(const uv4*)(W + (size_t)(n0 + r) * Pd + col * 8);
  }
  __syncthreads();

  const int w    = tid >> 5;          // wave id 0..7 (wave32)
  const int lane = tid & 31;
  const int wm   = w & 3;             // 4 waves along M
  const int wn   = w >> 2;            // 2 waves along N
  const int rl   = lane & 15;
  const int kh   = (lane < 16) ? 0 : 8;   // ISA 16-bit A layout: lanes 0-15 K{0..7,16..23}, 16-31 K{8..15,24..31}

  union Frag { uv4 q[2]; v16bf v; };
  v8f zero = {0.f, 0.f, 0.f, 0.f, 0.f, 0.f, 0.f, 0.f};
  v8f acc[2][4];
  #pragma unroll
  for (int i = 0; i < 2; ++i)
    #pragma unroll
    for (int j = 0; j < 4; ++j) acc[i][j] = zero;

  #pragma unroll
  for (int kc = 0; kc < 4; ++kc) {
    const int k0 = kc * 32;
    Frag af[2], bf[4];
    #pragma unroll
    for (int mt = 0; mt < 2; ++mt) {
      const ushort_t* src = sA + (wm * 32 + mt * 16 + rl) * 136 + k0 + kh;
      af[mt].q[0] = *(const uv4*)(src);
      af[mt].q[1] = *(const uv4*)(src + 16);
    }
    #pragma unroll
    for (int nt = 0; nt < 4; ++nt) {
      const ushort_t* src = sB + (wn * 64 + nt * 16 + rl) * 136 + k0 + kh;
      bf[nt].q[0] = *(const uv4*)(src);
      bf[nt].q[1] = *(const uv4*)(src + 16);
    }
    #pragma unroll
    for (int mt = 0; mt < 2; ++mt)
      #pragma unroll
      for (int nt = 0; nt < 4; ++nt)
        acc[mt][nt] = __builtin_amdgcn_wmma_f32_16x16x32_bf16(
            false, af[mt].v, false, bf[nt].v, (short)0, acc[mt][nt], false, false);
  }

  // epilogue: C layout — lane L: N = L%16; VGPR i -> M = i + (L<16 ? 0 : 8)
  #pragma unroll
  for (int nt = 0; nt < 4; ++nt) {
    const int ng = n0 + wn * 64 + nt * 16 + rl;
    const float bias = (ng < N_OBS) ? obs_b[ng] : st_b[ng - N_OBS];
    #pragma unroll
    for (int mt = 0; mt < 2; ++mt) {
      #pragma unroll
      for (int i = 0; i < 8; ++i) {
        const int mg = m0 + wm * 32 + mt * 16 + i + kh;
        const float v = acc[mt][nt][i] + bias;
        if (ng < N_OBS) out_obs[(size_t)mg * N_OBS + ng] = v;
        else            out_st [(size_t)mg * N_ST  + (ng - N_OBS)] = v;
      }
    }
  }
}

// =====================================================================
// Kernel 3a: per-(b,t) loss partials (deterministic fixed-order reductions)
// =====================================================================
__global__ void __launch_bounds__(256)
teml_loss_part(const float* __restrict__ ol, const float* __restrict__ sl,
               const int* __restrict__ obs, const int* __restrict__ st_idx,
               const float* __restrict__ p, const float* __restrict__ pm,
               float* __restrict__ partial) {
  __shared__ float red[256];
  const int tid = threadIdx.x;
  const int m   = blockIdx.x;        // b*T + t
  const int t   = m & (Tv - 1);

  // obs log-softmax NLL (512 logits, 2 per thread)
  const float* orow = ol + (size_t)m * N_OBS;
  float v0 = orow[tid], v1 = orow[tid + 256];
  float mx  = redMax(fmaxf(v0, v1), red, tid, 256);
  float se  = redSum(__expf(v0 - mx) + __expf(v1 - mx), red, tid, 256);
  float lse = logf(se) + mx;
  float obs_nll = lse - orow[obs[m]];

  // state log-softmax NLL (1024 logits, 4 per thread)
  const float* srow = sl + (size_t)m * N_ST;
  float a0 = srow[tid], a1 = srow[tid + 256], a2 = srow[tid + 512], a3 = srow[tid + 768];
  float mx2  = redMax(fmaxf(fmaxf(a0, a1), fmaxf(a2, a3)), red, tid, 256);
  float se2  = redSum(__expf(a0 - mx2) + __expf(a1 - mx2) + __expf(a2 - mx2) + __expf(a3 - mx2),
                      red, tid, 256);
  float lse2 = logf(se2) + mx2;
  float st_nll = lse2 - srow[st_idx[m]];

  // mem term: (p_mem[b,t] - p[b,t-1])^2 for t >= 1
  float msq = 0.0f;
  if (t > 0 && tid < Pd) {
    float dd = pm[(size_t)m * Pd + tid] - p[(size_t)(m - 1) * Pd + tid];
    msq = dd * dd;
  }
  float memsum = redSum(msq, red, tid, 256);

  if (tid == 0) {
    const float invBT  = 1.0f / (float)BT;
    const float invMem = 1.0f / ((float)Bv * (float)(Tv - 1) * (float)Pd);
    partial[m] = OBS_W * obs_nll * invBT + ST_W * st_nll * invBT + MEM_W * memsum * invMem;
  }
}

// Kernel 3b: fixed-order final reduction of 65536 partials -> scalar loss
__global__ void __launch_bounds__(256)
teml_loss_final(const float* __restrict__ partial, float* __restrict__ loss) {
  __shared__ float red[256];
  const int tid = threadIdx.x;
  float s = 0.0f;
  for (int i = tid; i < BT; i += 256) s += partial[i];   // fixed strided order
  float tot = redSum(s, red, tid, 256);
  if (tid == 0) loss[0] = tot;
}

// =====================================================================
extern "C" void kernel_launch(void* const* d_in, const int* in_sizes, int n_in,
                              void* d_out, int out_size, void* d_ws, size_t ws_size,
                              hipStream_t stream) {
  (void)in_sizes; (void)n_in; (void)out_size; (void)ws_size;

  const int*   obs      = (const int*)  d_in[0];
  const int*   actions  = (const int*)  d_in[1];
  const int*   states   = (const int*)  d_in[2];
  const float* obs_emb  = (const float*)d_in[3];
  const float* act_emb  = (const float*)d_in[4];
  const float* g_init   = (const float*)d_in[5];
  const float* w_ih     = (const float*)d_in[6];
  const float* w_hh     = (const float*)d_in[7];
  const float* b_ih     = (const float*)d_in[8];
  const float* b_hh     = (const float*)d_in[9];
  const float* enc_w1   = (const float*)d_in[10];
  const float* enc_b1   = (const float*)d_in[11];
  const float* ln_g     = (const float*)d_in[12];
  const float* ln_b     = (const float*)d_in[13];
  const float* enc_w2   = (const float*)d_in[14];
  const float* enc_b2   = (const float*)d_in[15];
  const float* dec_ow   = (const float*)d_in[16];
  const float* dec_ob   = (const float*)d_in[17];
  const float* dec_sw   = (const float*)d_in[18];
  const float* dec_sb   = (const float*)d_in[19];

  float* out = (float*)d_out;
  float* out_obsl = out + OFF_OBSL;
  float* out_stl  = out + OFF_STL;
  float* out_p    = out + OFF_P;
  float* out_pm   = out + OFF_PM;
  float* out_loss = out + OFF_LOSS;

  // workspace layout
  ushort_t* pcomb   = (ushort_t*)d_ws;                                  // BT*128 bf16 = 16 MB
  ushort_t* Wbf     = (ushort_t*)((char*)d_ws + (size_t)BT * Pd * 2);   // 1536*128 bf16
  float*    partial = (float*)((char*)d_ws + (size_t)BT * Pd * 2 + (size_t)N_TOT * Pd * 2);

  // 0) pack decoder weights to bf16
  teml_prep_w<<<(N_TOT * Pd) / 256, 256, 0, stream>>>(dec_ow, dec_sw, Wbf);

  // 1) recurrent phase: 1 WG per batch element, M resident in LDS (~72.3 KB dyn LDS)
  size_t lds1 = (size_t)(128 * MSTR + 64 + 64 + 32 + 192 + 192 + 128 * 5) * sizeof(float);
  teml_recurrent<<<Bv, 128, lds1, stream>>>(
      obs, actions, obs_emb, act_emb, g_init, w_ih, w_hh, b_ih, b_hh,
      enc_w1, enc_b1, ln_g, ln_b, enc_w2, enc_b2, out_p, out_pm, pcomb);

  // 2) decoder GEMM with WMMA (bf16 x bf16 -> f32)
  size_t lds2 = (size_t)2 * 128 * 136 * sizeof(ushort_t);               // 69,632 B
  teml_decoder_gemm<<<dim3(BT / 128, N_TOT / 128), 256, lds2, stream>>>(
      pcomb, Wbf, dec_ob, dec_sb, out_obsl, out_stl);

  // 3) loss
  teml_loss_part<<<BT, 256, 0, stream>>>(out_obsl, out_stl, obs, states, out_p, out_pm, partial);
  teml_loss_final<<<1, 256, 0, stream>>>(partial, out_loss);
}
